// MultiHeadAttention_9826885173936
// MI455X (gfx1250) — matrix operand order repaired and checked by reading, not compiled, not run on hardware
//
#include <hip/hip_runtime.h>
#include <hip/hip_bf16.h>

// ---------------------------------------------------------------------------
// Cross-attention transformer block for MI455X (gfx1250, wave32, WMMA).
// All GEMM contractions run on v_wmma_f32_16x16x32_f16 (f16 in, f32 acc).
// GEMM: async (GLOBAL_LOAD_ASYNC_TO_LDS) double-buffered weight tiles,
//       32x64 C tile per wave, B fragments preloaded one step ahead.
// Attention: flash-style streaming softmax per (batch, head, 16-row tile),
//       V tiles staged to LDS with async copies overlapped with score WMMAs.
// ---------------------------------------------------------------------------

typedef _Float16 h8v  __attribute__((ext_vector_type(8)));
typedef _Float16 v16h __attribute__((ext_vector_type(16)));
typedef float    v8f  __attribute__((ext_vector_type(8)));

// Problem constants (match the reference file exactly).
#define TX   12276
#define TM   12276
#define TXP  12288   // padded to multiple of 256
#define TMP  12288
#define DMODEL 512
#define NHEAD  8
#define DHEAD  64
#define DFF    2048

// LDS weight tile: 64 cols(C) x 64 K, rows padded to 72 f16 for bank spread.
#define WT_K    64
#define WT_PAD  72

__constant__ int c_LX[16]   = {1024,896,768,640,512,1000,750,600,1024,512,700,800,900,950,650,550};
__constant__ int c_XOFF[16] = {0,1024,1920,2688,3328,3840,4840,5590,6190,7214,7726,8426,9226,10126,11076,11726};
__constant__ int c_LM[16]   = {550,650,950,900,800,700,512,1024,600,750,1000,512,640,768,896,1024};
__constant__ int c_MOFF[16] = {0,550,1200,2150,3050,3850,4550,5062,6086,6686,7436,8436,8948,9588,10356,11252};

static __device__ __forceinline__ v16h frag2(const _Float16* p0, const _Float16* p1) {
    h8v lo = *(const h8v*)p0;
    h8v hi = *(const h8v*)p1;
    return __builtin_shufflevector(lo, hi, 0,1,2,3,4,5,6,7,8,9,10,11,12,13,14,15);
}

static __device__ __forceinline__ v8f wmma_f16(v16h a, v16h b, v8f c) {
    return __builtin_amdgcn_wmma_f32_16x16x32_f16(
        /*neg_a=*/false, a, /*neg_b=*/false, b,
        /*c_mod=*/(short)0, c, /*reuse_a=*/false, /*reuse_b=*/false);
}

// CDNA5 async global->LDS copy (16B per lane), tracked by ASYNCcnt.
// ldsOff = LDS byte offset (low 32 bits of a generic shared pointer).
static __device__ __forceinline__ void async_ld_b128(unsigned ldsOff, const _Float16* g) {
    asm volatile("global_load_async_to_lds_b128 %0, %1, off"
                 :: "v"(ldsOff), "v"(g) : "memory");
}
static __device__ __forceinline__ void wait_async0() {
    asm volatile("s_wait_asynccnt 0x0" ::: "memory");
}
static __device__ __forceinline__ unsigned lds_off(const void* p) {
    return (unsigned)(uintptr_t)p;   // generic LDS pointer: low 32 bits = LDS offset
}

// ---------------------------------------------------------------------------
// Elementwise helpers
// ---------------------------------------------------------------------------
__global__ __launch_bounds__(256) void cvt_f32_f16(const float* __restrict__ s,
                                                   _Float16* __restrict__ d, int n) {
    int i = blockIdx.x * 256 + threadIdx.x;
    if (i < n) d[i] = (_Float16)s[i];
}

// Convert with row padding: rows >= validRows written as zero.
__global__ __launch_bounds__(256) void cvt_pad_f16(const float* __restrict__ s,
                                                   _Float16* __restrict__ d,
                                                   int validRows, int padRows, int cols) {
    int i = blockIdx.x * 256 + threadIdx.x;
    if (i >= padRows * cols) return;
    int r = i / cols;
    d[i] = (r < validRows) ? (_Float16)s[i] : (_Float16)0.0f;
}

__global__ __launch_bounds__(256) void zero_f16(_Float16* __restrict__ d, int n) {
    int i = blockIdx.x * 256 + threadIdx.x;
    if (i < n) d[i] = (_Float16)0.0f;
}

// out = x1 + (float)f2   (final residual, fp32 output)
__global__ __launch_bounds__(256) void final_add(const float* __restrict__ x1,
                                                 const _Float16* __restrict__ f2,
                                                 float* __restrict__ out, int n) {
    int i = blockIdx.x * 256 + threadIdx.x;
    if (i < n) out[i] = x1[i] + (float)f2[i];
}

// ---------------------------------------------------------------------------
// LayerNorm: one block per row (512 cols, 256 threads x 2 elems) -> f16 out
// ---------------------------------------------------------------------------
__global__ __launch_bounds__(256) void ln_kernel(const float* __restrict__ x,
                                                 const float* __restrict__ g,
                                                 const float* __restrict__ b,
                                                 _Float16* __restrict__ out, int rows) {
    int row = blockIdx.x;
    if (row >= rows) return;
    int t = threadIdx.x;
    const float* xr = x + (size_t)row * DMODEL;
    float v0 = xr[t], v1 = xr[t + 256];
    __shared__ float red[256];
    red[t] = v0 + v1;
    __syncthreads();
    for (int s = 128; s > 0; s >>= 1) { if (t < s) red[t] += red[t + s]; __syncthreads(); }
    float mean = red[0] * (1.0f / DMODEL);
    __syncthreads();
    float d0 = v0 - mean, d1 = v1 - mean;
    red[t] = d0 * d0 + d1 * d1;
    __syncthreads();
    for (int s = 128; s > 0; s >>= 1) { if (t < s) red[t] += red[t + s]; __syncthreads(); }
    float rs = rsqrtf(red[0] * (1.0f / DMODEL) + 1e-5f);
    _Float16* orow = out + (size_t)row * DMODEL;
    orow[t]       = (_Float16)(d0 * rs * g[t]       + b[t]);
    orow[t + 256] = (_Float16)(d1 * rs * g[t + 256] + b[t + 256]);
}

// x1 = x + (float)ao ; h2 = LN(x1) (fp32 residual kept, f16 normalized out)
__global__ __launch_bounds__(256) void res_ln_kernel(const float* __restrict__ x,
                                                     const _Float16* __restrict__ ao,
                                                     const float* __restrict__ g,
                                                     const float* __restrict__ b,
                                                     float* __restrict__ x1,
                                                     _Float16* __restrict__ h2, int rows) {
    int row = blockIdx.x;
    if (row >= rows) return;
    int t = threadIdx.x;
    size_t base = (size_t)row * DMODEL;
    float v0 = x[base + t]       + (float)ao[base + t];
    float v1 = x[base + t + 256] + (float)ao[base + t + 256];
    x1[base + t] = v0;
    x1[base + t + 256] = v1;
    __shared__ float red[256];
    red[t] = v0 + v1;
    __syncthreads();
    for (int s = 128; s > 0; s >>= 1) { if (t < s) red[t] += red[t + s]; __syncthreads(); }
    float mean = red[0] * (1.0f / DMODEL);
    __syncthreads();
    float d0 = v0 - mean, d1 = v1 - mean;
    red[t] = d0 * d0 + d1 * d1;
    __syncthreads();
    for (int s = 128; s > 0; s >>= 1) { if (t < s) red[t] += red[t + s]; __syncthreads(); }
    float rs = rsqrtf(red[0] * (1.0f / DMODEL) + 1e-5f);
    h2[base + t]       = (_Float16)(d0 * rs * g[t]       + b[t]);
    h2[base + t + 256] = (_Float16)(d1 * rs * g[t + 256] + b[t + 256]);
}

// ---------------------------------------------------------------------------
// WMMA GEMM: C[M,N](f16) = A[M,K](f16) * W[N,K](f16)^T (+bias, optional relu)
//
// Block = 8 waves = 256 rows x 64 cols of C. Each wave owns a 32x64 tile
// (8 f32x8 accumulators). Weight tile (64 cols x 64 K) is staged in LDS,
// double-buffered via CDNA5 async global->LDS copies: the next chunk's
// copy is issued before the current chunk's WMMAs, and completion costs one
// s_wait_asynccnt + one barrier per chunk. B fragments (ds_load_b128 pairs)
// are preloaded one step ahead so LDS latency hides under the matrix ops;
// each fragment feeds two WMMAs (two 16-row groups).
//
// Requires: M % 256 == 0, N % 64 == 0, K % 64 == 0.
// grid.x = (M/256) * (N/64)
// ---------------------------------------------------------------------------
__global__ __launch_bounds__(256) void gemm_wmma(const _Float16* __restrict__ A,
                                                 const _Float16* __restrict__ W,
                                                 _Float16* __restrict__ C,
                                                 const float* __restrict__ bias,
                                                 int M, int N, int K, int relu) {
    int wave  = threadIdx.x >> 5;
    int lane  = threadIdx.x & 31;
    int lhalf = lane >> 4;        // 0..1
    int lmod  = lane & 15;        // 0..15
    int nblk64 = N >> 6;
    int mblk = blockIdx.x / nblk64;
    int nblk = blockIdx.x % nblk64;
    int mbase = mblk * 256 + wave * 32;
    int nbase = nblk * 64;

    __shared__ alignas(16) _Float16 Wlds[2][64 * WT_PAD];

    // Cooperative stage of one 64x64 weight chunk: 512 16B pieces / 256 thr.
    const int cidx0 = threadIdx.x;          // piece 0
    const int cidx1 = threadIdx.x + 256;    // piece 1
    const int r0 = cidx0 >> 3, c0 = (cidx0 & 7) * 8;
    const int r1 = cidx1 >> 3, c1 = (cidx1 & 7) * 8;
    const _Float16* Wr0 = W + (size_t)(nbase + r0) * K + c0;
    const _Float16* Wr1 = W + (size_t)(nbase + r1) * K + c1;
    const unsigned ldsBase = lds_off(&Wlds[0][0]);
    const unsigned off0 = (unsigned)(r0 * WT_PAD + c0) * 2u;
    const unsigned off1 = (unsigned)(r1 * WT_PAD + c1) * 2u;
    const unsigned bufBytes = 64 * WT_PAD * 2u;

    // A row pointers: two 16-row groups per wave.
    const _Float16* ArA = A + (size_t)(mbase + lmod) * K;
    const _Float16* ArB = A + (size_t)(mbase + 16 + lmod) * K;

    v8f acc[2][4] = {};

    // Prime buffer 0 with async copy.
    async_ld_b128(ldsBase + off0, Wr0);
    async_ld_b128(ldsBase + off1, Wr1);
    wait_async0();
    __syncthreads();

    const int nch = K / WT_K;
    int cur = 0;
    for (int ch = 0; ch < nch; ++ch) {
        const int kc = ch * WT_K;
        const bool have_next = (ch + 1) < nch;

        // Kick off async copy of the NEXT weight chunk into the other buffer;
        // it proceeds in the background while this chunk's WMMAs execute.
        if (have_next) {
            unsigned nb = ldsBase + (cur ^ 1) * bufBytes;
            async_ld_b128(nb + off0, Wr0 + kc + WT_K);
            async_ld_b128(nb + off1, Wr1 + kc + WT_K);
        }

        // Issue all A fragments for this chunk up front (8 global b128 loads).
        v16h a0k0 = frag2(ArA + kc      + lhalf * 8, ArA + kc      + lhalf * 8 + 16);
        v16h a1k0 = frag2(ArB + kc      + lhalf * 8, ArB + kc      + lhalf * 8 + 16);
        v16h a0k1 = frag2(ArA + kc + 32 + lhalf * 8, ArA + kc + 32 + lhalf * 8 + 16);
        v16h a1k1 = frag2(ArB + kc + 32 + lhalf * 8, ArB + kc + 32 + lhalf * 8 + 16);
        if (have_next) __builtin_prefetch(ArA + kc + WT_K, 0, 3);

        // 8 B fragments (idx = ks*4 + c), preloaded one step ahead.
        const _Float16* Wb = &Wlds[0][cur * (64 * WT_PAD)];
        const _Float16* fp0 = Wb + lmod * WT_PAD + lhalf * 8;   // c=0, ks=0
        v16h bf = frag2(fp0, fp0 + 16);
#pragma unroll
        for (int idx = 0; idx < 8; ++idx) {
            v16h bnext = bf;
            if (idx < 7) {
                int nidx = idx + 1;
                const _Float16* bp = Wb + ((nidx & 3) * 16 + lmod) * WT_PAD
                                        + (nidx >> 2) * 32 + lhalf * 8;
                bnext = frag2(bp, bp + 16);
            }
            v16h aa0 = (idx >= 4) ? a0k1 : a0k0;
            v16h aa1 = (idx >= 4) ? a1k1 : a1k0;
            acc[0][idx & 3] = wmma_f16(aa0, bf, acc[0][idx & 3]);
            acc[1][idx & 3] = wmma_f16(aa1, bf, acc[1][idx & 3]);
            bf = bnext;
        }

        if (have_next) {
            wait_async0();      // next chunk resident in LDS
            __syncthreads();    // one barrier per chunk
            cur ^= 1;
        }
    }

    // Epilogue: C layout row = i + (lane/16)*8 within each 16-row group.
#pragma unroll
    for (int r = 0; r < 2; ++r) {
#pragma unroll
        for (int c = 0; c < 4; ++c) {
            int col = nbase + c * 16 + lmod;
            float bv = bias ? bias[col] : 0.0f;
#pragma unroll
            for (int i = 0; i < 8; ++i) {
                int row = mbase + r * 16 + i + lhalf * 8;
                float v = acc[r][c][i] + bv;
                if (relu && v < 0.0f) v = 0.0f;
                C[(size_t)row * N + col] = (_Float16)v;
            }
        }
    }
}

// ---------------------------------------------------------------------------
// Flash-style masked cross-attention. One wave per (batch, head, 16-row xtile).
// Streams memory positions in chunks of 32; online softmax; WMMA for both
// q*k^T and P*V. V tiles are staged to LDS with async copies that complete
// under the score WMMAs + softmax; P is bounced through LDS to convert its
// C-layout into an A-fragment layout.
// grid.x = 16 * 8 * 64 ; block = 32
// ---------------------------------------------------------------------------
__global__ __launch_bounds__(32) void attn_kernel(const _Float16* __restrict__ q,
                                                  const _Float16* __restrict__ k,
                                                  const _Float16* __restrict__ v,
                                                  _Float16* __restrict__ ao) {
    int bid = blockIdx.x;
    int b   = bid >> 9;          // NHEAD*64 = 512 blocks per batch
    int rem = bid & 511;
    int h   = rem >> 6;
    int xt  = rem & 63;
    int Lx = c_LX[b], Lm = c_LM[b];
    if (xt * 16 >= Lx) return;

    int lane  = threadIdx.x;
    int lhalf = lane >> 4;
    int lmod  = lane & 15;

    // q fragments (A of the score WMMA): row x = lane%16 of this tile.
    int qrow = c_XOFF[b] + xt * 16 + lmod;
    if (qrow >= TXP) qrow = TXP - 1;
    const _Float16* qp = q + (size_t)qrow * DMODEL + h * DHEAD;
    v16h aq0 = frag2(qp + lhalf * 8,      qp + lhalf * 8 + 16);       // dh  0..31
    v16h aq1 = frag2(qp + 32 + lhalf * 8, qp + 32 + lhalf * 8 + 16);  // dh 32..63

    float m[8], l[8];
#pragma unroll
    for (int i = 0; i < 8; ++i) { m[i] = -1e30f; l[i] = 0.0f; }
    v8f O[4] = {};

    __shared__ alignas(16) _Float16 Plds[16 * 32];
    __shared__ alignas(16) _Float16 Vlds[32 * 64];
    const unsigned vldsBase = lds_off(&Vlds[0]);

    int nyc = (Lm + 31) / 32;
    for (int yc = 0; yc < nyc; ++yc) {
        int ybase = yc * 32;
        __syncthreads();  // protect LDS WAR vs previous iteration

        // Async-stage V tile (32 y-rows x 64 dh): one row per lane, 8x16B.
        // Copy proceeds while the score WMMAs and softmax run below.
        {
            int yr = ybase + lane;
            int gy = c_MOFF[b] + (yr < Lm ? yr : Lm - 1);
            const _Float16* vp = v + (size_t)gy * DMODEL + h * DHEAD;
            unsigned dst = vldsBase + (unsigned)lane * 128u;
#pragma unroll
            for (int c = 0; c < 8; ++c)
                async_ld_b128(dst + c * 16u, vp + c * 8);
        }

        // Scores: two 16x16 subtiles, each accumulated over dh in 2 WMMA steps.
        v8f S[2] = {};
#pragma unroll
        for (int ys = 0; ys < 2; ++ys) {
            int y = ybase + ys * 16 + lmod;
            int gy = c_MOFF[b] + (y < Lm ? y : Lm - 1);
            const _Float16* kp = k + (size_t)gy * DMODEL + h * DHEAD + lhalf * 8;
            v16h bk0 = frag2(kp,      kp + 16);   // dh  0..31
            v16h bk1 = frag2(kp + 32, kp + 48);   // dh 32..63
            S[ys] = wmma_f16(aq0, bk0, S[ys]);
            S[ys] = wmma_f16(aq1, bk1, S[ys]);
        }

        // Online softmax (C-layout rows live across 16-lane halves).
        bool mask0 = (ybase + lmod)      >= Lm;
        bool mask1 = (ybase + 16 + lmod) >= Lm;
        float p0[8], p1[8];
#pragma unroll
        for (int i = 0; i < 8; ++i) {
            float s0 = mask0 ? -1e9f : S[0][i] * 0.125f;   // 1/sqrt(64)
            float s1 = mask1 ? -1e9f : S[1][i] * 0.125f;
            float cm = fmaxf(s0, s1);
            cm = fmaxf(cm, __shfl_xor(cm, 8));
            cm = fmaxf(cm, __shfl_xor(cm, 4));
            cm = fmaxf(cm, __shfl_xor(cm, 2));
            cm = fmaxf(cm, __shfl_xor(cm, 1));
            float nm = fmaxf(m[i], cm);
            float sc = __expf(m[i] - nm);
            float e0 = __expf(s0 - nm);
            float e1 = __expf(s1 - nm);
            float rs = e0 + e1;
            rs += __shfl_xor(rs, 8);
            rs += __shfl_xor(rs, 4);
            rs += __shfl_xor(rs, 2);
            rs += __shfl_xor(rs, 1);
            l[i] = l[i] * sc + rs;
            m[i] = nm;
#pragma unroll
            for (int c = 0; c < 4; ++c) O[c][i] *= sc;
            p0[i] = e0; p1[i] = e1;
        }

        // Bounce P through LDS: C-layout -> A-fragment layout.
#pragma unroll
        for (int i = 0; i < 8; ++i) {
            int prow = i + lhalf * 8;
            Plds[prow * 32 + lmod]      = (_Float16)p0[i];
            Plds[prow * 32 + 16 + lmod] = (_Float16)p1[i];
        }
        wait_async0();     // V tile resident in LDS
        __syncthreads();

        const _Float16* pr = Plds + lmod * 32 + lhalf * 16;
        v16h aP = frag2(pr, pr + 8);

#pragma unroll
        for (int c = 0; c < 4; ++c) {
            v16h bV = {};
#pragma unroll
            for (int e = 0; e < 16; ++e)
                bV[e] = Vlds[(lhalf * 16 + e) * 64 + c * 16 + lmod];
            O[c] = wmma_f16(aP, bV, O[c]);
        }
    }

    // Normalize and write concat-head output (only valid x rows).
#pragma unroll
    for (int i = 0; i < 8; ++i) {
        int xr = xt * 16 + i + lhalf * 8;
        if (xr >= Lx) continue;
        size_t row = (size_t)(c_XOFF[b] + xr);
        float inv = 1.0f / l[i];
#pragma unroll
        for (int c = 0; c < 4; ++c)
            ao[row * DMODEL + h * DHEAD + c * 16 + lmod] = (_Float16)(O[c][i] * inv);
    }
}

// ---------------------------------------------------------------------------
// Host-side launch
// ---------------------------------------------------------------------------
extern "C" void kernel_launch(void* const* d_in, const int* in_sizes, int n_in,
                              void* d_out, int out_size, void* d_ws, size_t ws_size,
                              hipStream_t stream) {
    (void)in_sizes; (void)n_in; (void)out_size; (void)ws_size;
    const float* x     = (const float*)d_in[0];
    const float* mem   = (const float*)d_in[1];
    const float* Wq    = (const float*)d_in[2];
    const float* Wk    = (const float*)d_in[3];
    const float* Wv    = (const float*)d_in[4];
    const float* Wo    = (const float*)d_in[5];
    const float* W1    = (const float*)d_in[6];
    const float* b1    = (const float*)d_in[7];
    const float* W2    = (const float*)d_in[8];
    const float* b2    = (const float*)d_in[9];
    const float* g_in  = (const float*)d_in[10];
    const float* b_in  = (const float*)d_in[11];
    const float* g_it  = (const float*)d_in[12];
    const float* b_it  = (const float*)d_in[13];
    float* out = (float*)d_out;

    // Workspace carve (256B aligned slabs).
    char* p = (char*)d_ws;
    auto alloc = [&](size_t bytes) -> void* {
        void* r = (void*)p;
        p += (bytes + 255) & ~(size_t)255;
        return r;
    };
    _Float16* Wq16 = (_Float16*)alloc((size_t)DMODEL * DMODEL * 2);
    _Float16* Wk16 = (_Float16*)alloc((size_t)DMODEL * DMODEL * 2);
    _Float16* Wv16 = (_Float16*)alloc((size_t)DMODEL * DMODEL * 2);
    _Float16* Wo16 = (_Float16*)alloc((size_t)DMODEL * DMODEL * 2);
    _Float16* W116 = (_Float16*)alloc((size_t)DFF * DMODEL * 2);
    _Float16* W216 = (_Float16*)alloc((size_t)DMODEL * DFF * 2);
    _Float16* h16   = (_Float16*)alloc((size_t)TXP * DMODEL * 2);
    _Float16* mem16 = (_Float16*)alloc((size_t)TMP * DMODEL * 2);
    _Float16* q16   = (_Float16*)alloc((size_t)TXP * DMODEL * 2);
    _Float16* k16   = (_Float16*)alloc((size_t)TMP * DMODEL * 2);
    _Float16* v16   = (_Float16*)alloc((size_t)TMP * DMODEL * 2);
    _Float16* ac16  = (_Float16*)alloc((size_t)TXP * DMODEL * 2);  // concat-head attn
    _Float16* ao16  = (_Float16*)alloc((size_t)TXP * DMODEL * 2);  // attn @ Wo^T
    _Float16* h216  = (_Float16*)alloc((size_t)TXP * DMODEL * 2);
    _Float16* f116  = (_Float16*)alloc((size_t)TXP * DFF * 2);
    _Float16* f216  = (_Float16*)alloc((size_t)TXP * DMODEL * 2);
    float*    x1    = (float*)alloc((size_t)TX * DMODEL * 4);

    const int SQ = DMODEL * DMODEL;   // 262144
    const int SF = DFF * DMODEL;      // 1048576

    // Weight conversion.
    cvt_f32_f16<<<(SQ + 255) / 256, 256, 0, stream>>>(Wq, Wq16, SQ);
    cvt_f32_f16<<<(SQ + 255) / 256, 256, 0, stream>>>(Wk, Wk16, SQ);
    cvt_f32_f16<<<(SQ + 255) / 256, 256, 0, stream>>>(Wv, Wv16, SQ);
    cvt_f32_f16<<<(SQ + 255) / 256, 256, 0, stream>>>(Wo, Wo16, SQ);
    cvt_f32_f16<<<(SF + 255) / 256, 256, 0, stream>>>(W1, W116, SF);
    cvt_f32_f16<<<(SF + 255) / 256, 256, 0, stream>>>(W2, W216, SF);

    // mem -> f16 with zero pad rows.
    cvt_pad_f16<<<((TMP * DMODEL) + 255) / 256, 256, 0, stream>>>(mem, mem16, TM, TMP, DMODEL);

    // LayerNorm(x) -> h16; zero pad rows.
    ln_kernel<<<TX, 256, 0, stream>>>(x, g_in, b_in, h16, TX);
    zero_f16<<<(((TXP - TX) * DMODEL) + 255) / 256, 256, 0, stream>>>(
        h16 + (size_t)TX * DMODEL, (TXP - TX) * DMODEL);

    // Projections (M padded to 12288 = 48*256, N=512, K=512).
    const int gProj = (TXP / 256) * (DMODEL / 64);   // 48 * 8  = 384
    gemm_wmma<<<gProj, 256, 0, stream>>>(h16,   Wq16, q16, nullptr, TXP, DMODEL, DMODEL, 0);
    gemm_wmma<<<gProj, 256, 0, stream>>>(mem16, Wk16, k16, nullptr, TMP, DMODEL, DMODEL, 0);
    gemm_wmma<<<gProj, 256, 0, stream>>>(mem16, Wv16, v16, nullptr, TMP, DMODEL, DMODEL, 0);

    // Attention: one wave per (batch, head, 16-row x tile).
    attn_kernel<<<16 * NHEAD * 64, 32, 0, stream>>>(q16, k16, v16, ac16);

    // Output projection.
    gemm_wmma<<<gProj, 256, 0, stream>>>(ac16, Wo16, ao16, nullptr, TXP, DMODEL, DMODEL, 0);

    // Residual + second LayerNorm; zero pad rows of h2.
    res_ln_kernel<<<TX, 256, 0, stream>>>(x, ao16, g_it, b_it, x1, h216, TX);
    zero_f16<<<(((TXP - TX) * DMODEL) + 255) / 256, 256, 0, stream>>>(
        h216 + (size_t)TX * DMODEL, (TXP - TX) * DMODEL);

    // FFN: relu(h2 @ W1^T + b1) @ W2^T + b2
    const int gF1 = (TXP / 256) * (DFF / 64);        // 48 * 32 = 1536
    gemm_wmma<<<gF1, 256, 0, stream>>>(h216, W116, f116, b1, TXP, DFF, DMODEL, 1);
    gemm_wmma<<<gProj, 256, 0, stream>>>(f116, W216, f216, b2, TXP, DMODEL, DFF, 0);

    // Final residual in fp32 -> d_out.
    final_add<<<((TX * DMODEL) + 255) / 256, 256, 0, stream>>>(x1, f216, out, TX * DMODEL);
}
